// IGNN_Layer_81088982548660
// MI455X (gfx1250) — matrix operand order repaired and checked by reading, not compiled, not run on hardware
//
#include <hip/hip_runtime.h>
#include <stdint.h>

typedef __attribute__((ext_vector_type(16))) __bf16 v16bf;
typedef __attribute__((ext_vector_type(8)))  __bf16 v8bf;
typedef __attribute__((ext_vector_type(8)))  float  v8f;
typedef int v4i __attribute__((vector_size(16)));

#define P_DIM 128
#define N_DIM 4096
#define K_ENC 512
#define KAPPA_F 0.9f
#define ITERS 50
#define KSPLIT 4
#define KSLICE (N_DIM / KSPLIT)   // 1024

#if defined(__has_builtin)
#if __has_builtin(__builtin_amdgcn_global_load_async_to_lds_b128) && \
    __has_builtin(__builtin_amdgcn_s_wait_asynccnt)
#define HAS_ASYNC_LDS 1
#endif
#endif
#ifndef HAS_ASYNC_LDS
#define HAS_ASYNC_LDS 0
#endif

// float -> bf16 round-to-nearest-even
__device__ __forceinline__ unsigned short f2bf_rne(float f) {
    unsigned u = __float_as_uint(f);
    u += 0x7FFFu + ((u >> 16) & 1u);
    return (unsigned short)(u >> 16);
}

// 16B global -> LDS copy: async path on gfx1250 (ASYNCcnt), sync fallback.
__device__ __forceinline__ void cp16_g2l(unsigned short* lds_dst, const unsigned short* gsrc) {
#if HAS_ASYNC_LDS
    __builtin_amdgcn_global_load_async_to_lds_b128(
        (__attribute__((address_space(1))) v4i*)(void*)const_cast<unsigned short*>(gsrc),
        (__attribute__((address_space(3))) v4i*)(void*)lds_dst,
        0, 0);
#else
    *reinterpret_cast<v8bf*>(lds_dst) = *reinterpret_cast<const v8bf*>(gsrc);
#endif
}
__device__ __forceinline__ void async_wait_all() {
#if HAS_ASYNC_LDS
    __builtin_amdgcn_s_wait_asynccnt(0);
#endif
}

// A-fragment: 16x32 bf16 from row-major A.
// lanes 0-15: row m=lane, elems 0..7 = K0..7, 8..15 = K16..23
// lanes 16-31: row m=lane-16, elems 0..7 = K8..15, 8..15 = K24..31
__device__ __forceinline__ v16bf load_a_frag(const unsigned short* __restrict__ A,
                                             int lda, int m0, int k0, int lane) {
    int m  = m0 + (lane & 15);
    int kk = k0 + 8 * (lane >> 4);
    v8bf a0 = *reinterpret_cast<const v8bf*>(A + (size_t)m * lda + kk);
    v8bf a1 = *reinterpret_cast<const v8bf*>(A + (size_t)m * lda + kk + 16);
    return __builtin_shufflevector(a0, a1, 0,1,2,3,4,5,6,7,8,9,10,11,12,13,14,15);
}

// ---------------- proj_norm_inf: bisection replaces sort (same fixpoint) ---------
__global__ void __launch_bounds__(128) proj_kernel(const float* __restrict__ W,
                                                   unsigned short* __restrict__ WpBF) {
    __shared__ float red[128];
    const int r = blockIdx.x;
    const int t = threadIdx.x;
    float w = W[r * 128 + t];
    float a = fabsf(w);

    red[t] = a; __syncthreads();
    for (int s = 64; s > 0; s >>= 1) { if (t < s) red[t] += red[t + s]; __syncthreads(); }
    float rowsum = red[0]; __syncthreads();

    red[t] = a; __syncthreads();
    for (int s = 64; s > 0; s >>= 1) { if (t < s) red[t] = fmaxf(red[t], red[t + s]); __syncthreads(); }
    float amax = red[0]; __syncthreads();

    float theta = 0.0f;
    if (rowsum > KAPPA_F) {      // uniform across block
        float lo = 0.0f, hi = amax;
        for (int it = 0; it < 48; ++it) {
            float mid = 0.5f * (lo + hi);
            red[t] = fmaxf(a - mid, 0.0f); __syncthreads();
            for (int s = 64; s > 0; s >>= 1) { if (t < s) red[t] += red[t + s]; __syncthreads(); }
            float sm = red[0]; __syncthreads();
            if (sm > KAPPA_F) lo = mid; else hi = mid;
        }
        theta = 0.5f * (lo + hi);
    }
    float wp = (rowsum > KAPPA_F) ? copysignf(fmaxf(a - theta, 0.0f), w) : w;
    WpBF[r * 128 + t] = f2bf_rne(wp);
}

// ---------------- B = (F0 @ W_enc^T + b_enc)^T, f32 (one-time) -------------------
__global__ void __launch_bounds__(128) encode_kernel(const float* __restrict__ F0,
                                                     const float* __restrict__ Wenc,
                                                     const float* __restrict__ benc,
                                                     float* __restrict__ Bm) {
    __shared__ float f0s[K_ENC];
    const int n = blockIdx.x;
    const int p = threadIdx.x;
    for (int k = p; k < K_ENC; k += 128) f0s[k] = F0[(size_t)n * K_ENC + k];
    __syncthreads();
    float s = 0.0f;
    for (int k = 0; k < K_ENC; ++k) s = fmaf(f0s[k], Wenc[(size_t)p * K_ENC + k], s);
    Bm[(size_t)p * N_DIM + n] = s + benc[p];
}

// ---------------- f32 -> bf16 conversion (grid-stride) ---------------------------
__global__ void convert_f32_bf16(const float* __restrict__ src,
                                 unsigned short* __restrict__ dst, int n) {
    int i = blockIdx.x * blockDim.x + threadIdx.x;
    int stride = gridDim.x * blockDim.x;
    for (; i < n; i += stride) dst[i] = f2bf_rne(src[i]);
}

// ---------------- T = Wp @ X  (128x128 x 128x4096, bf16 WMMA) --------------------
// 64 WGs, 64-column tiles; whole 128x64 X tile staged once in LDS (async), shared
// by all 8 waves.
__global__ void __launch_bounds__(256) gemm1_WpX(const unsigned short* __restrict__ Wp,
                                                 const unsigned short* __restrict__ X,
                                                 unsigned short* __restrict__ T) {
    __shared__ __align__(32) unsigned short Xs[128 * 64];   // 16 KB
    const int lane  = threadIdx.x & 31;
    const int wave  = threadIdx.x >> 5;
    const int nbase = blockIdx.x * 64;
    const int m0    = wave * 16;

    // stage: 256 threads x 64B (4 x b128)
    {
        int row = threadIdx.x >> 1;          // 0..127 (K row)
        int col = (threadIdx.x & 1) * 32;    // 0 or 32
        #pragma unroll
        for (int c = 0; c < 4; ++c)
            cp16_g2l(&Xs[row * 64 + col + c * 8],
                     &X[(size_t)row * N_DIM + nbase + col + c * 8]);
    }
    async_wait_all();
    __syncthreads();

    v8f z = {0.f,0.f,0.f,0.f,0.f,0.f,0.f,0.f};
    v8f acc[4];
    for (int t = 0; t < 4; ++t) acc[t] = z;

    #pragma unroll
    for (int k0 = 0; k0 < 128; k0 += 32) {
        v16bf a = load_a_frag(Wp, 128, m0, k0, lane);
        #pragma unroll
        for (int t = 0; t < 4; ++t) {
            v16bf b;
            #pragma unroll
            for (int s = 0; s < 16; ++s)
                b[s] = __builtin_bit_cast(__bf16, Xs[(k0 + 8 * (lane >> 4) + (s & 7) + 16 * (s >> 3)) * 64 + t * 16 + (lane & 15)]);
            acc[t] = __builtin_amdgcn_wmma_f32_16x16x32_bf16(
                         false, a, false, b, (short)0, acc[t], false, false);
        }
    }
    const int half = lane >> 4;
    const int ncol = lane & 15;
    for (int t = 0; t < 4; ++t) {
        int n = nbase + t * 16 + ncol;
        #pragma unroll
        for (int r = 0; r < 8; ++r) {
            int m = m0 + half * 8 + r;
            T[(size_t)m * N_DIM + n] = f2bf_rne(acc[t][r]);
        }
    }
}

// ---- P[kslice] = T[:, kbeg:kend] @ H[kbeg:kend, ntile]  (f32 partials) ----------
// grid (64 n-tiles, KSPLIT k-slices) = 256 WGs. H slab (32x64 bf16 = 4KB) double-
// buffered in LDS, streamed with async global->LDS loads.
__global__ void __launch_bounds__(256) gemm2_partial(const unsigned short* __restrict__ T,
                                                     const unsigned short* __restrict__ Hbf,
                                                     float* __restrict__ Pbase) {
    __shared__ __align__(32) unsigned short Hs[2][32 * 64];
    const int lane  = threadIdx.x & 31;
    const int wave  = threadIdx.x >> 5;
    const int nbase = blockIdx.x * 64;
    const int kbeg  = blockIdx.y * KSLICE;
    const int kend  = kbeg + KSLICE;
    float* Pout = Pbase + (size_t)blockIdx.y * P_DIM * N_DIM;
    const int m0 = wave * 16;

    const int srow = threadIdx.x >> 3;        // 0..31 (K row in slab)
    const int scol = (threadIdx.x & 7) * 8;   // 8 elems = 16B per thread

    v8f z = {0.f,0.f,0.f,0.f,0.f,0.f,0.f,0.f};
    v8f acc[4];
    for (int t = 0; t < 4; ++t) acc[t] = z;

    // prologue stage of first slab
    cp16_g2l(&Hs[0][srow * 64 + scol],
             &Hbf[(size_t)(kbeg + srow) * N_DIM + nbase + scol]);

    int cur = 0;
    for (int k0 = kbeg; k0 < kend; k0 += 32) {
        async_wait_all();
        __syncthreads();                       // slab `cur` ready for everyone
        if (k0 + 32 < kend)                    // prefetch next slab into other buffer
            cp16_g2l(&Hs[cur ^ 1][srow * 64 + scol],
                     &Hbf[(size_t)(k0 + 32 + srow) * N_DIM + nbase + scol]);

        v16bf a = load_a_frag(T, N_DIM, m0, k0, lane);
        #pragma unroll
        for (int t = 0; t < 4; ++t) {
            v16bf b;
            #pragma unroll
            for (int s = 0; s < 16; ++s)
                b[s] = __builtin_bit_cast(__bf16, Hs[cur][(8 * (lane >> 4) + (s & 7) + 16 * (s >> 3)) * 64 + t * 16 + (lane & 15)]);
            acc[t] = __builtin_amdgcn_wmma_f32_16x16x32_bf16(
                         false, a, false, b, (short)0, acc[t], false, false);
        }
        __syncthreads();                       // reads of `cur` done before it is rewritten
        cur ^= 1;
    }

    const int half = lane >> 4;
    const int ncol = lane & 15;
    for (int t = 0; t < 4; ++t) {
        int n = nbase + t * 16 + ncol;
        #pragma unroll
        for (int r = 0; r < 8; ++r) {
            int m = m0 + half * 8 + r;
            Pout[(size_t)m * N_DIM + n] = acc[t][r];
        }
    }
}

// ---- Y = relu(sum_k P_k + B); write bf16 X_next and optionally f32 B + Y --------
__global__ void __launch_bounds__(256) combine_kernel(const float* __restrict__ Pbase,
                                                      const float* __restrict__ Bm,
                                                      unsigned short* __restrict__ Xout,
                                                      float* __restrict__ outF) {
    const size_t sz = (size_t)P_DIM * N_DIM;
    size_t i = (size_t)blockIdx.x * blockDim.x + threadIdx.x;
    float bv = Bm[i];
    float s = bv;
    #pragma unroll
    for (int k = 0; k < KSPLIT; ++k) s += Pbase[k * sz + i];
    float y = s > 0.0f ? s : 0.0f;
    Xout[i] = f2bf_rne(y);
    if (outF) outF[i] = y + bv;                // B + X*
}

extern "C" void kernel_launch(void* const* d_in, const int* in_sizes, int n_in,
                              void* d_out, int out_size, void* d_ws, size_t ws_size,
                              hipStream_t stream) {
    const float* X0   = (const float*)d_in[0];   // (128, 4096)
    const float* H    = (const float*)d_in[1];   // (4096, 4096)
    const float* F0   = (const float*)d_in[3];   // (4096, 512)
    const float* W    = (const float*)d_in[4];   // (128, 128)
    const float* Wenc = (const float*)d_in[5];   // (128, 512)
    const float* benc = (const float*)d_in[6];   // (128,)
    float* out = (float*)d_out;                  // (128, 4096)

    char* ws = (char*)d_ws;
    unsigned short* Hbf  = (unsigned short*)ws;  ws += (size_t)N_DIM * N_DIM * 2;          // 32 MB
    float*          Bm   = (float*)ws;           ws += (size_t)P_DIM * N_DIM * 4;          //  2 MB
    float*          Pp   = (float*)ws;           ws += (size_t)KSPLIT * P_DIM * N_DIM * 4; //  8 MB
    unsigned short* Tbf  = (unsigned short*)ws;  ws += (size_t)P_DIM * N_DIM * 2;          //  1 MB
    unsigned short* Xa   = (unsigned short*)ws;  ws += (size_t)P_DIM * N_DIM * 2;          //  1 MB
    unsigned short* Xb   = (unsigned short*)ws;  ws += (size_t)P_DIM * N_DIM * 2;          //  1 MB
    unsigned short* WpBF = (unsigned short*)ws;  ws += (size_t)128 * 128 * 2;

    proj_kernel  <<<128,  128, 0, stream>>>(W, WpBF);
    encode_kernel<<<N_DIM,128, 0, stream>>>(F0, Wenc, benc, Bm);
    convert_f32_bf16<<<2048, 256, 0, stream>>>(H,  Hbf, N_DIM * N_DIM);
    convert_f32_bf16<<<512,  256, 0, stream>>>(X0, Xa,  P_DIM * N_DIM);

    unsigned short* Xcur = Xa;
    unsigned short* Xnxt = Xb;
    const int nelem = P_DIM * N_DIM;
    for (int it = 0; it < ITERS; ++it) {
        gemm1_WpX<<<64, 256, 0, stream>>>(WpBF, Xcur, Tbf);
        gemm2_partial<<<dim3(64, KSPLIT), 256, 0, stream>>>(Tbf, Hbf, Pp);
        float* of = (it == ITERS - 1) ? out : nullptr;
        combine_kernel<<<nelem / 256, 256, 0, stream>>>(Pp, Bm, Xnxt, of);
        unsigned short* tmp = Xcur; Xcur = Xnxt; Xnxt = tmp;
    }
}
